// VectorQuantizerEMA_89945205113202
// MI455X (gfx1250) — matrix operand order repaired and checked by reading, not compiled, not run on hardware
//
#include <hip/hip_runtime.h>
#include <hip/hip_bf16.h>
#include <stdint.h>

// ---------------------------------------------------------------------------
// VQ-VAE EMA vector quantizer, MI455X (gfx1250, wave32, WMMA).
// Distance GEMM runs on v_wmma_f32_16x16x32_bf16 with fused argmin.
// Round 3: M=32 per wave (B fragments feed two independent accumulation
// chains -> half the DS traffic / waits per WMMA) + LDS double-buffering
// (global staging loads overlap the WMMA chain; one barrier per tile).
// ---------------------------------------------------------------------------

typedef __attribute__((ext_vector_type(16))) __bf16 v16bf;
typedef __attribute__((ext_vector_type(8)))  float  v8f;

#define N_VEC   32768
#define C_DIM   256
#define K_CODES 1024
#define DECAY_F 0.99f
#define EPS_F   1e-5f

// ---------------------------------------------------------------------------
// e_norm[k] = sum_c E[c,k]^2
__global__ __launch_bounds__(256) void vq_enorm(const float* __restrict__ E,
                                                float* __restrict__ enorm) {
  int k = blockIdx.x * 256 + threadIdx.x;
  if (k < K_CODES) {
    float s = 0.f;
    for (int c = 0; c < C_DIM; ++c) {
      float v = E[(size_t)c * K_CODES + k];
      s += v * v;
    }
    enorm[k] = s;
  }
}

// ---------------------------------------------------------------------------
// zero scratch accumulators (ws is poisoned by harness)
__global__ __launch_bounds__(256) void vq_init(float* __restrict__ cs,
                                               float* __restrict__ dw,
                                               float* __restrict__ loss_acc) {
  int i = blockIdx.x * 256 + threadIdx.x;
  if (i < K_CODES) cs[i] = 0.f;
  if (i < C_DIM * K_CODES) dw[i] = 0.f;
  if (i == 0) loss_acc[0] = 0.f;
}

// ---------------------------------------------------------------------------
// Main kernel: argmin_k ( e_norm[k] - 2 * x_n . e_k )  via bf16 WMMA.
// Each wave owns TWO 16-row tiles (32 rows); workgroup = 8 waves = 256 rows.
// Loop over 64 tiles of 16 codes; E tiles double-buffered (bf16) in LDS.
__global__ __launch_bounds__(256) void vq_argmin_wmma(
    const float* __restrict__ x,       // [N, 256] row-major
    const float* __restrict__ E,       // [256, 1024]
    const float* __restrict__ enorm,   // [1024]
    int* __restrict__ idx_out) {

  __shared__ __bf16 Btile[2][16 * C_DIM];   // [buf][code][c], 16 KB

  const int tid  = threadIdx.x;
  const int lane = tid & 31;
  const int wave = tid >> 5;
  const int m0   = (blockIdx.x * 8 + wave) * 32;

  const int m   = lane & 15;                 // A: row within tile
  const int kgA = (lane < 16) ? 0 : 16;      // A: K-group offset (contiguous, matches B)
  const int nB  = lane & 15;                 // B: column (code) within tile
  const int ksB = (lane < 16) ? 0 : 16;      // B: K-group offset (contiguous, matches A)

  // --- Preload A fragments for both row tiles (32 rows x 256 dims, bf16).
  // A 16x32 layout (matches B): lane holds K = kg..kg+15 contiguous.
  v16bf afrag[2][8];
#pragma unroll
  for (int t = 0; t < 2; ++t) {
    const float* xrow = x + (size_t)(m0 + t * 16 + m) * C_DIM;
#pragma unroll
    for (int cc = 0; cc < 8; ++cc) {
      const int cb = cc * 32;
#pragma unroll
      for (int e = 0; e < 8; ++e) {
        afrag[t][cc][e]     = (__bf16)xrow[cb + kgA + e];
        afrag[t][cc][e + 8] = (__bf16)xrow[cb + kgA + 8 + e];
      }
    }
  }

  float minv[2][8];
  int   mini[2][8];
#pragma unroll
  for (int t = 0; t < 2; ++t)
#pragma unroll
    for (int r = 0; r < 8; ++r) { minv[t][r] = 3.0e38f; mini[t][r] = 0; }

  // Staging ownership: thread = (code tk, 16-wide c segment tseg).
  // Consecutive tids share c and read consecutive codes -> 64B coalesced.
  const int tk   = tid & 15;
  const int tseg = tid >> 4;

  // Prologue: stage tile 0 into buffer 0.
  {
    v16bf bst;
#pragma unroll
    for (int e = 0; e < 16; ++e)
      bst[e] = (__bf16)E[(size_t)(tseg * 16 + e) * K_CODES + tk];
    *(v16bf*)&Btile[0][tk * C_DIM + tseg * 16] = bst;   // 2x ds_store_b128
  }
  __syncthreads();

  for (int kt = 0; kt < K_CODES / 16; ++kt) {
    const int k0  = kt * 16;
    const int buf = kt & 1;

    // Issue next tile's global staging loads FIRST (longest latency),
    // then the B-fragment DS loads; the WMMA chain only gates on DScnt.
    float sreg[16];
    if (kt + 1 < K_CODES / 16) {
      const int k0n = k0 + 16;
#pragma unroll
      for (int e = 0; e < 16; ++e)
        sreg[e] = E[(size_t)(tseg * 16 + e) * K_CODES + k0n + tk];
      __builtin_prefetch(&E[(size_t)tid * K_CODES + k0n + 16], 0, 1);
    }

    // B fragments (32x16, same order as A: lane K = ks..ks+15 contiguous).
    v16bf bfrag[8];
#pragma unroll
    for (int cc = 0; cc < 8; ++cc)
      bfrag[cc] = *(const v16bf*)&Btile[buf][nB * C_DIM + cc * 32 + ksB];

    // Two independent accumulation chains share each B fragment.
    v8f acc0 = {};
    v8f acc1 = {};
#pragma unroll
    for (int cc = 0; cc < 8; ++cc) {
      acc0 = __builtin_amdgcn_wmma_f32_16x16x32_bf16(
          false, afrag[0][cc], false, bfrag[cc], (short)0, acc0, false, false);
      acc1 = __builtin_amdgcn_wmma_f32_16x16x32_bf16(
          false, afrag[1][cc], false, bfrag[cc], (short)0, acc1, false, false);
    }

    // Convert + store next tile into the other buffer (overlaps WMMA drain).
    if (kt + 1 < K_CODES / 16) {
      v16bf bst;
#pragma unroll
      for (int e = 0; e < 16; ++e) bst[e] = (__bf16)sreg[e];
      *(v16bf*)&Btile[buf ^ 1][tk * C_DIM + tseg * 16] = bst;
    }

    const float en  = enorm[k0 + nB];
    const int   col = k0 + nB;
#pragma unroll
    for (int r = 0; r < 8; ++r) {
      const float s0 = en - 2.0f * acc0[r];  // row-constant ||x||^2 dropped
      const float s1 = en - 2.0f * acc1[r];
      if (s0 < minv[0][r]) { minv[0][r] = s0; mini[0][r] = col; }
      if (s1 < minv[1][r]) { minv[1][r] = s1; mini[1][r] = col; }
    }
    // One barrier per tile: guards both "next buffer staged" and
    // "current buffer fully consumed before it is re-staged at kt+1".
    __syncthreads();
  }

  // Reduce (min, first-index) across the 16 lanes sharing each row.
#pragma unroll
  for (int t = 0; t < 2; ++t) {
#pragma unroll
    for (int r = 0; r < 8; ++r) {
#pragma unroll
      for (int off = 8; off >= 1; off >>= 1) {
        const float ov = __shfl_xor(minv[t][r], off, 16);
        const int   oi = __shfl_xor(mini[t][r], off, 16);
        if (ov < minv[t][r] || (ov == minv[t][r] && oi < mini[t][r])) {
          minv[t][r] = ov; mini[t][r] = oi;
        }
      }
    }
  }
  if ((lane & 15) == 0) {
    // C/D layout: VGPR r -> row r (lanes 0-15) / row 8+r (lanes 16-31).
    const int half = (lane < 16) ? 0 : 8;
#pragma unroll
    for (int t = 0; t < 2; ++t) {
      const int rowbase = m0 + t * 16 + half;
#pragma unroll
      for (int r = 0; r < 8; ++r) idx_out[rowbase + r] = mini[t][r];
    }
  }
}

// ---------------------------------------------------------------------------
// quant gather (quant_st forward value == quant) + commitment-loss partial sum
__global__ __launch_bounds__(256) void vq_gather(
    const float* __restrict__ x, const float* __restrict__ E,
    const int* __restrict__ idx, float* __restrict__ quant_out,
    float* __restrict__ loss_acc) {
  const int n = blockIdx.x;            // one row per block
  const int c = threadIdx.x;
  const size_t gid = (size_t)n * C_DIM + c;
  const int k = idx[n];
  const float q = E[(size_t)c * K_CODES + k];
  quant_out[gid] = q;
  const float d = q - x[gid];
  __shared__ float red[256];
  red[c] = d * d;
  __syncthreads();
  for (int o = 128; o > 0; o >>= 1) {
    if (c < o) red[c] += red[c + o];
    __syncthreads();
  }
  if (c == 0) unsafeAtomicAdd(loss_acc, red[0]);
}

// ---------------------------------------------------------------------------
// cluster-size histogram
__global__ __launch_bounds__(256) void vq_hist(const int* __restrict__ idx,
                                               float* __restrict__ cs) {
  const int n = blockIdx.x * 256 + threadIdx.x;
  if (n < N_VEC) unsafeAtomicAdd(&cs[idx[n]], 1.0f);
}

// ---------------------------------------------------------------------------
// dw[c, idx[n]] += x[n, c]   (scatter-add form of flat^T @ one_hot)
__global__ __launch_bounds__(256) void vq_dw(const float* __restrict__ x,
                                             const int* __restrict__ idx,
                                             float* __restrict__ dw) {
  const int n = blockIdx.x;
  const int c = threadIdx.x;
  const int k = idx[n];
  unsafeAtomicAdd(&dw[(size_t)c * K_CODES + k], x[(size_t)n * C_DIM + c]);
}

// ---------------------------------------------------------------------------
// new_ema_cs, n = sum(new_ema_cs), perplexity, commit_loss  (one workgroup)
__global__ __launch_bounds__(1024) void vq_fin_cs(
    const float* __restrict__ ema_cs, const float* __restrict__ cs,
    float* __restrict__ out_ncs, float* __restrict__ n_tot,
    const float* __restrict__ loss_acc, float* __restrict__ out_loss,
    float* __restrict__ out_perp) {
  const int k = threadIdx.x;
  const float ncs = ema_cs[k] * DECAY_F + cs[k] * (1.0f - DECAY_F);
  out_ncs[k] = ncs;
  const float p  = cs[k] * (1.0f / (float)N_VEC);     // avg_probs
  const float pl = p * __logf(p + 1e-10f);
  __shared__ float r1[1024];
  __shared__ float r2[1024];
  r1[k] = ncs; r2[k] = pl;
  __syncthreads();
  for (int o = 512; o > 0; o >>= 1) {
    if (k < o) { r1[k] += r1[k + o]; r2[k] += r2[k + o]; }
    __syncthreads();
  }
  if (k == 0) {
    n_tot[0]    = r1[0];
    out_perp[0] = __expf(-r2[0]);
    out_loss[0] = 0.25f * loss_acc[0] / (float)((size_t)N_VEC * C_DIM);
  }
}

// ---------------------------------------------------------------------------
// new_ema_dw and normalized embeddings
__global__ __launch_bounds__(256) void vq_emb(
    const float* __restrict__ ema_dw, const float* __restrict__ dw,
    const float* __restrict__ ncs, const float* __restrict__ n_tot,
    float* __restrict__ out_ndw, float* __restrict__ out_emb) {
  const int gid = blockIdx.x * 256 + threadIdx.x;
  if (gid >= C_DIM * K_CODES) return;
  const int k = gid & (K_CODES - 1);
  const float ndw = ema_dw[gid] * DECAY_F + dw[gid] * (1.0f - DECAY_F);
  out_ndw[gid] = ndw;
  const float n  = n_tot[0];
  const float sm = (ncs[k] + EPS_F) / (n + (float)K_CODES * EPS_F) * n;
  out_emb[gid] = ndw / sm;
}

// ---------------------------------------------------------------------------
extern "C" void kernel_launch(void* const* d_in, const int* in_sizes, int n_in,
                              void* d_out, int out_size, void* d_ws, size_t ws_size,
                              hipStream_t stream) {
  const float* x      = (const float*)d_in[0];   // [32,32,32,256]
  const float* E      = (const float*)d_in[1];   // [256,1024]
  const float* ema_cs = (const float*)d_in[2];   // [1024]
  const float* ema_dw = (const float*)d_in[3];   // [256,1024]

  float* out = (float*)d_out;
  float* out_quant = out;                                   // 8388608
  float* out_emb   = out_quant + (size_t)N_VEC * C_DIM;     // 262144
  float* out_ncs   = out_emb + (size_t)C_DIM * K_CODES;     // 1024
  float* out_ndw   = out_ncs + K_CODES;                     // 262144
  float* out_loss  = out_ndw + (size_t)C_DIM * K_CODES;     // 1
  float* out_perp  = out_loss + 1;                          // 1

  float* ws       = (float*)d_ws;
  float* enorm    = ws;                         // 1024
  int*   idx      = (int*)(ws + K_CODES);       // 32768
  float* cs       = ws + K_CODES + N_VEC;       // 1024
  float* dw       = cs + K_CODES;               // 262144
  float* loss_acc = dw + (size_t)C_DIM * K_CODES;
  float* n_tot    = loss_acc + 1;

  vq_enorm<<<(K_CODES + 255) / 256, 256, 0, stream>>>(E, enorm);
  vq_init<<<(C_DIM * K_CODES + 255) / 256, 256, 0, stream>>>(cs, dw, loss_acc);
  vq_argmin_wmma<<<N_VEC / 256, 256, 0, stream>>>(x, E, enorm, idx);
  vq_gather<<<N_VEC, 256, 0, stream>>>(x, E, idx, out_quant, loss_acc);
  vq_hist<<<N_VEC / 256, 256, 0, stream>>>(idx, cs);
  vq_dw<<<N_VEC, 256, 0, stream>>>(x, idx, dw);
  vq_fin_cs<<<1, 1024, 0, stream>>>(ema_cs, cs, out_ncs, n_tot, loss_acc,
                                    out_loss, out_perp);
  vq_emb<<<(C_DIM * K_CODES + 255) / 256, 256, 0, stream>>>(
      ema_dw, dw, out_ncs, n_tot, out_ndw, out_emb);
}
